// Decoder_73100343377961
// MI455X (gfx1250) — matrix operand order repaired and checked by reading, not compiled, not run on hardware
//
#include <hip/hip_runtime.h>
#include <cstddef>
#include <cstdint>

// Problem constants (from reference): B=128, T=256, DX=256, H=512, S=512
constexpr int BB  = 128;
constexpr int TT  = 256;
constexpr int DXX = 256;
constexpr int HH  = 512;
constexpr int SS  = 512;
constexpr int KC  = DXX + 2 * HH + HH;   // 1792 : [x_t | h_ctx | c_ctx | h]
constexpr int N4H = 4 * HH;              // 2048

typedef __attribute__((ext_vector_type(16))) __bf16 bf16x16;
typedef __attribute__((ext_vector_type(8)))  float  f32x8;

__device__ __forceinline__ float sigmoidf_(float x) {
  return 1.0f / (1.0f + __expf(-x));
}

// ---------------- init kernels (run once per launch; deterministic) --------

// Pack Wc[n][k] = k < 1280 ? W_ih[n][k] : W_hh[n][k-1280], as bf16.
__global__ void k_pack_weights(const float* __restrict__ W_ih,
                               const float* __restrict__ W_hh,
                               __bf16* __restrict__ Wc) {
  int idx = blockIdx.x * blockDim.x + threadIdx.x;
  if (idx >= N4H * KC) return;
  int n = idx / KC;
  int k = idx - n * KC;
  float v = (k < DXX + 2 * HH)
                ? W_ih[(size_t)n * (DXX + 2 * HH) + k]
                : W_hh[(size_t)n * HH + (k - (DXX + 2 * HH))];
  Wc[idx] = (__bf16)v;
}

// h <- h0, c <- c0, bias <- b_ih + b_hh
__global__ void k_init_state(const float* __restrict__ h0,
                             const float* __restrict__ c0,
                             const float* __restrict__ b_ih,
                             const float* __restrict__ b_hh,
                             float* __restrict__ h, float* __restrict__ c,
                             float* __restrict__ bias) {
  int idx = blockIdx.x * blockDim.x + threadIdx.x;
  if (idx < BB * HH) { h[idx] = h0[idx]; c[idx] = c0[idx]; }
  if (idx < N4H) bias[idx] = b_ih[idx] + b_hh[idx];
}

// ---------------- fused dual attention + input assembly --------------------
// One block per batch row. Computes BOTH h-attention and c-attention with a
// single pass over enc for scores and a single pass for contexts (halves the
// dominant L2 traffic vs separate attends). Writes the full bf16 GEMM input
// row [x_t | h_ctx | c_ctx | h]. 256 threads.
__global__ void __launch_bounds__(256)
k_attend(const float* __restrict__ x,    // (B,T,DX)
         const float* __restrict__ enc,  // (B,S,H)
         const float* __restrict__ h,    // (B,H)
         const float* __restrict__ c,    // (B,H)
         __bf16* __restrict__ inp,       // (B,KC) bf16
         int t) {
  __shared__ float qh[HH];
  __shared__ float qc[HH];
  __shared__ float sch[SS];   // h scores -> exp values
  __shared__ float scc[SS];   // c scores -> exp values
  __shared__ float2 red[256];

  const int b   = blockIdx.x;
  const int tid = threadIdx.x;

  const float* hp = h + (size_t)b * HH;
  const float* cp = c + (size_t)b * HH;
  qh[tid]       = hp[tid];
  qh[tid + 256] = hp[tid + 256];
  qc[tid]       = cp[tid];
  qc[tid + 256] = cp[tid + 256];
  __syncthreads();

  const float* encb = enc + (size_t)b * SS * HH;

  // --- scores for both queries in ONE pass over enc ---
  float vh[2], vc[2];
  float mh = -3.4e38f, mc = -3.4e38f;
#pragma unroll
  for (int r = 0; r < 2; ++r) {
    const int s = tid + r * 256;
    const float4* e4  = reinterpret_cast<const float4*>(encb + (size_t)s * HH);
    const float4* qh4 = reinterpret_cast<const float4*>(qh);
    const float4* qc4 = reinterpret_cast<const float4*>(qc);
    float ah = 0.f, ac = 0.f;
#pragma unroll 4
    for (int k = 0; k < HH / 4; ++k) {
      float4 e = e4[k];
      float4 a = qh4[k];
      float4 d = qc4[k];
      ah += e.x * a.x + e.y * a.y + e.z * a.z + e.w * a.w;
      ac += e.x * d.x + e.y * d.y + e.z * d.z + e.w * d.w;
    }
    vh[r] = ah; vc[r] = ac;
    mh = fmaxf(mh, ah); mc = fmaxf(mc, ac);
  }

  // --- paired max reduction (h and c together) ---
  red[tid] = make_float2(mh, mc);
  __syncthreads();
  for (int off = 128; off > 0; off >>= 1) {
    if (tid < off) {
      float2 o = red[tid + off];
      float2 m = red[tid];
      red[tid] = make_float2(fmaxf(m.x, o.x), fmaxf(m.y, o.y));
    }
    __syncthreads();
  }
  const float mxh = red[0].x;
  const float mxc = red[0].y;
  __syncthreads();

  // --- exp + paired sum reduction ---
  const float eh0 = __expf(vh[0] - mxh);
  const float eh1 = __expf(vh[1] - mxh);
  const float ec0 = __expf(vc[0] - mxc);
  const float ec1 = __expf(vc[1] - mxc);
  sch[tid]       = eh0;
  sch[tid + 256] = eh1;
  scc[tid]       = ec0;
  scc[tid + 256] = ec1;
  red[tid] = make_float2(eh0 + eh1, ec0 + ec1);
  __syncthreads();
  for (int off = 128; off > 0; off >>= 1) {
    if (tid < off) {
      float2 o = red[tid + off];
      float2 m = red[tid];
      red[tid] = make_float2(m.x + o.x, m.y + o.y);
    }
    __syncthreads();
  }
  const float invh = 1.0f / red[0].x;
  const float invc = 1.0f / red[0].y;

  // --- both contexts in ONE pass over enc ---
  // thread owns output elements hh = tid and tid+256
  float ch0 = 0.f, ch1 = 0.f, cc0 = 0.f, cc1 = 0.f;
  for (int s = 0; s < SS; ++s) {
    const float ah = sch[s] * invh;
    const float ac = scc[s] * invc;
    const float* er = encb + (size_t)s * HH;
    const float e0 = er[tid];
    const float e1 = er[tid + 256];
    ch0 += ah * e0; ch1 += ah * e1;
    cc0 += ac * e0; cc1 += ac * e1;
  }

  __bf16* ib = inp + (size_t)b * KC;
  // h_ctx at [DXX, DXX+HH)
  ib[DXX + tid]       = (__bf16)ch0;
  ib[DXX + tid + 256] = (__bf16)ch1;
  // c_ctx at [DXX+HH, DXX+2H)
  ib[DXX + HH + tid]       = (__bf16)cc0;
  ib[DXX + HH + tid + 256] = (__bf16)cc1;
  // x_t at [0, DXX)
  ib[tid] = (__bf16)x[((size_t)b * TT + t) * DXX + tid];
  // h at [DXX+2H, KC)
  ib[DXX + 2 * HH + tid]       = (__bf16)qh[tid];
  ib[DXX + 2 * HH + tid + 256] = (__bf16)qh[tid + 256];
}

// ---------------- fused gate GEMM: gates = inp_full @ Wc.T + bias ----------
// M=128, N=2048, K=1792, bf16 A/B, f32 accumulate.
// Block = 128 threads (4 waves); each wave owns a 16(M) x 32(N) tile
// (2 accumulators) -> grid (M/16, N/128) = (8, 16) = 128 blocks / 512 waves
// for better WGP coverage of this latency-bound, L2-hot GEMM.
__global__ void __launch_bounds__(128)
k_gates_wmma(const __bf16* __restrict__ inp,   // (B, KC)  row-major
             const __bf16* __restrict__ Wc,    // (N4H, KC) row-major (= B^T)
             const float* __restrict__ bias,   // (N4H)
             float* __restrict__ gates) {      // (B, N4H)
  const int lane = threadIdx.x & 31;
  const int wave = threadIdx.x >> 5;
  const int hl   = lane >> 4;   // lane half (0/1)
  const int lr   = lane & 15;   // lane row within half
  const int mtile = blockIdx.x * 16;
  const int nbase = (blockIdx.y * 4 + wave) * 32;

  f32x8 acc[2] = {};

  const __bf16* arow = inp + (size_t)(mtile + lr) * KC;
  for (int k0 = 0; k0 < KC; k0 += 32) {
    // A fragment (16x32 bf16). Layout matching B: lanes 0-15 hold K = 0..15,
    // lanes 16-31 hold K = 16..31, all for row M = lane%16.
    bf16x16 a;
    {
      const __bf16* p0 = arow + k0 + hl * 16;  // 8 contiguous bf16 (16B)
      const __bf16* p1 = p0 + 8;               // next 8 contiguous bf16
#pragma unroll
      for (int i = 0; i < 8; ++i) { a[i] = p0[i]; a[8 + i] = p1[i]; }
    }
    // 2 B fragments (32x16 bf16 each). Contiguous k order, as A: lanes 0-15 hold K=0..15,
    // lanes 16-31 hold K=16..31, for column N = lane%16. B[k][n] = Wc[n][k].
#pragma unroll
    for (int j = 0; j < 2; ++j) {
      bf16x16 bf;
      const __bf16* pb = Wc + (size_t)(nbase + j * 16 + lr) * KC + k0 + hl * 16;
#pragma unroll
      for (int i = 0; i < 16; ++i) bf[i] = pb[i];
      acc[j] = __builtin_amdgcn_wmma_f32_16x16x32_bf16(
          false, a, false, bf, (short)0, acc[j], false, false);
    }
  }

  // C layout: VGPR r holds (m = hl*8 + r, n = lane%16)
#pragma unroll
  for (int j = 0; j < 2; ++j) {
    const int ncol = nbase + j * 16 + lr;
    const float bv = bias[ncol];
#pragma unroll
    for (int r = 0; r < 8; ++r) {
      const int mrow = mtile + hl * 8 + r;
      gates[(size_t)mrow * N4H + ncol] = acc[j][r] + bv;
    }
  }
}

// ---------------- pointwise LSTM cell --------------------------------------
__global__ void __launch_bounds__(256)
k_lstm(const float* __restrict__ gates,  // (B, 4H) order [i|f|g|o]
       float* __restrict__ h, float* __restrict__ c,
       float* __restrict__ out,          // (B, T, H)
       int t) {
  const int idx = blockIdx.x * blockDim.x + threadIdx.x;  // < B*H
  const int b = idx >> 9;        // /512
  const int j = idx & (HH - 1);
  const float* g = gates + (size_t)b * N4H;
  const float gi = g[j];
  const float gf = g[HH + j];
  const float gg = g[2 * HH + j];
  const float go = g[3 * HH + j];
  const float c1 = sigmoidf_(gf) * c[idx] + sigmoidf_(gi) * tanhf(gg);
  const float h1 = sigmoidf_(go) * tanhf(c1);
  c[idx] = c1;
  h[idx] = h1;
  out[((size_t)b * TT + t) * HH + j] = h1;
}

// ---------------- launch ----------------------------------------------------
extern "C" void kernel_launch(void* const* d_in, const int* in_sizes, int n_in,
                              void* d_out, int out_size, void* d_ws, size_t ws_size,
                              hipStream_t stream) {
  (void)in_sizes; (void)n_in; (void)out_size; (void)ws_size;
  const float* x    = (const float*)d_in[0];
  const float* h0   = (const float*)d_in[1];
  const float* c0   = (const float*)d_in[2];
  const float* enc  = (const float*)d_in[3];
  const float* W_ih = (const float*)d_in[4];
  const float* W_hh = (const float*)d_in[5];
  const float* b_ih = (const float*)d_in[6];
  const float* b_hh = (const float*)d_in[7];
  float* out = (float*)d_out;

  // Workspace carve-out (~9.4 MB total)
  char* ws = (char*)d_ws;
  __bf16* Wc   = (__bf16*)ws; ws += (size_t)N4H * KC * sizeof(__bf16);
  __bf16* inp  = (__bf16*)ws; ws += (size_t)BB * KC * sizeof(__bf16);
  float* gates = (float*)ws;  ws += (size_t)BB * N4H * sizeof(float);
  float* h     = (float*)ws;  ws += (size_t)BB * HH * sizeof(float);
  float* c     = (float*)ws;  ws += (size_t)BB * HH * sizeof(float);
  float* bias  = (float*)ws;  ws += (size_t)N4H * sizeof(float);

  k_pack_weights<<<(N4H * KC + 255) / 256, 256, 0, stream>>>(W_ih, W_hh, Wc);
  k_init_state<<<(BB * HH + 255) / 256, 256, 0, stream>>>(h0, c0, b_ih, b_hh,
                                                          h, c, bias);

  for (int t = 0; t < TT; ++t) {
    k_attend<<<dim3(BB), 256, 0, stream>>>(x, enc, h, c, inp, t);
    k_gates_wmma<<<dim3(BB / 16, N4H / 128), 128, 0, stream>>>(inp, Wc, bias,
                                                               gates);
    k_lstm<<<(BB * HH) / 256, 256, 0, stream>>>(gates, h, c, out, t);
  }
}
